// PGD_46428596470394
// MI455X (gfx1250) — compile-verified, matrix operand order checked by reading, not run on hardware
//
#include <hip/hip_runtime.h>

typedef __attribute__((ext_vector_type(16))) _Float16 v16h;
typedef __attribute__((ext_vector_type(8)))  _Float16 v8h;
typedef __attribute__((ext_vector_type(8)))  float    v8f;

#define EPS_PGD  0.4f
#define NORM_EPS 1e-8f

constexpr int B_ = 8, S_ = 64, V_ = 50257, D_ = 512;
constexpr int T_   = B_ * S_;      // 512 tokens
constexpr int NPAD = 50688;        // 64 * 8 * 99 : vocab padded to 8-tile groups

// ---------------------------------------------------------------------------
// Pass 1: unit-normalize fp32 rows -> f16 rows (one wave32 per row).
// Rows in [rows_valid, rows_total) are zero padding.
// ---------------------------------------------------------------------------
__global__ void __launch_bounds__(256)
normalize_rows_f16(const float* __restrict__ src, _Float16* __restrict__ dst,
                   int rows_valid, int rows_total) {
  int wave = (int)((blockIdx.x * blockDim.x + threadIdx.x) >> 5);
  int lane = threadIdx.x & 31;
  if (wave >= rows_total) return;

  _Float16* drow = dst + (size_t)wave * D_;
  if (wave >= rows_valid) {                 // zero pad rows
    v8h z = {};
    *(v8h*)(drow + lane * 16)     = z;
    *(v8h*)(drow + lane * 16 + 8) = z;
    return;
  }

  const float* srow = src + (size_t)wave * D_;
  float vals[16];
  float ss = 0.f;
#pragma unroll
  for (int i = 0; i < 4; ++i) {
    float4 x = *(const float4*)(srow + lane * 16 + i * 4);
    vals[4*i+0] = x.x; vals[4*i+1] = x.y; vals[4*i+2] = x.z; vals[4*i+3] = x.w;
    ss += x.x*x.x + x.y*x.y + x.z*x.z + x.w*x.w;
  }
#pragma unroll
  for (int m = 16; m >= 1; m >>= 1) ss += __shfl_xor(ss, m, 32);
  float scale = 1.0f / fmaxf(sqrtf(ss), NORM_EPS);

  v8h lo, hi;
#pragma unroll
  for (int i = 0; i < 8; ++i) {
    lo[i] = (_Float16)(vals[i]     * scale);
    hi[i] = (_Float16)(vals[i + 8] * scale);
  }
  *(v8h*)(drow + lane * 16)     = lo;
  *(v8h*)(drow + lane * 16 + 8) = hi;
}

// ---------------------------------------------------------------------------
// Pass 2: perturbed = emb[utt] + eps*sign(grad); unit-normalize -> f16.
// One wave32 per token.
// ---------------------------------------------------------------------------
__global__ void __launch_bounds__(256)
perturb_normalize_f16(const int* __restrict__ utt, const float* __restrict__ emb,
                      const float* __restrict__ grad, _Float16* __restrict__ dst) {
  int wave = (int)((blockIdx.x * blockDim.x + threadIdx.x) >> 5);
  int lane = threadIdx.x & 31;
  if (wave >= T_) return;

  int tok = utt[wave];
  const float* erow = emb  + (size_t)tok  * D_;
  const float* grow = grad + (size_t)wave * D_;

  float vals[16];
  float ss = 0.f;
#pragma unroll
  for (int i = 0; i < 4; ++i) {
    float4 e = *(const float4*)(erow + lane * 16 + i * 4);
    float4 g = *(const float4*)(grow + lane * 16 + i * 4);
    float p0 = e.x + EPS_PGD * ((g.x > 0.f) ? 1.f : ((g.x < 0.f) ? -1.f : 0.f));
    float p1 = e.y + EPS_PGD * ((g.y > 0.f) ? 1.f : ((g.y < 0.f) ? -1.f : 0.f));
    float p2 = e.z + EPS_PGD * ((g.z > 0.f) ? 1.f : ((g.z < 0.f) ? -1.f : 0.f));
    float p3 = e.w + EPS_PGD * ((g.w > 0.f) ? 1.f : ((g.w < 0.f) ? -1.f : 0.f));
    vals[4*i+0] = p0; vals[4*i+1] = p1; vals[4*i+2] = p2; vals[4*i+3] = p3;
    ss += p0*p0 + p1*p1 + p2*p2 + p3*p3;
  }
#pragma unroll
  for (int m = 16; m >= 1; m >>= 1) ss += __shfl_xor(ss, m, 32);
  float scale = 1.0f / fmaxf(sqrtf(ss), NORM_EPS);

  _Float16* drow = dst + (size_t)wave * D_;
  v8h lo, hi;
#pragma unroll
  for (int i = 0; i < 8; ++i) {
    lo[i] = (_Float16)(vals[i]     * scale);
    hi[i] = (_Float16)(vals[i + 8] * scale);
  }
  *(v8h*)(drow + lane * 16)     = lo;
  *(v8h*)(drow + lane * 16 + 8) = hi;
}

// ---------------------------------------------------------------------------
// Pass 3: sim = Phat(512x512) * Ehat(NPADx512)^T via v_wmma_f32_16x16x32_f16.
//
// Block = 8 waves, all sharing one A tile (tile_m). A tile (16x512 f16, 16KB)
// is staged into LDS once per block with global_load_async_to_lds_b128 and
// read via ds_load_b128 (DS pipe), while B streams on the VMEM pipe.
// Each wave owns one 16(M) x 64(N) tile; K-loop is software-pipelined
// (next k-step's 5 fragments load into a second register set while the
// current 4 WMMAs issue) to break the load->wmma full-wait serialization.
//
// Wave32 16-bit fragment layout: lane<16 -> row=lane, halves K{k0..k0+7,
// k0+16..k0+23}; lane>=16 -> row=lane-16, halves K{k0+8..15, k0+24..31}.
// Row-major storage => A and B fragments load identically (no transpose).
// ---------------------------------------------------------------------------
union Frag16 { v16h v; v8h h[2]; };

__global__ void __launch_bounds__(256)
sim_gemm(const _Float16* __restrict__ phat, const _Float16* __restrict__ ehat,
         float* __restrict__ sim) {
  __shared__ _Float16 smemA[16 * D_];   // 16 KB: one M-tile of Phat

  int wid  = threadIdx.x >> 5;
  int lane = threadIdx.x & 31;
  int tile_m = blockIdx.x & 31;                    // 32 tiles over M=512
  int tile_n = (blockIdx.x >> 5) * 8 + wid;        // 792 tiles over NPAD

  // --- async stage of the shared A tile: 256 threads x 64 B ---
  {
    unsigned lds = (unsigned)(size_t)(&smemA[0]) + threadIdx.x * 64u;
    unsigned long long ga =
        (unsigned long long)(size_t)(phat + (size_t)tile_m * 16 * D_) +
        threadIdx.x * 64u;
#pragma unroll
    for (int i = 0; i < 4; ++i) {
      unsigned l = lds + i * 16u;
      unsigned long long g = ga + i * 16u;
      asm volatile("global_load_async_to_lds_b128 %0, %1, off"
                   :: "v"(l), "v"(g) : "memory");
    }
    asm volatile("s_wait_asynccnt 0x0" ::: "memory");
    __syncthreads();
  }

  int lrow = lane & 15;
  int koff = (lane < 16) ? 0 : 8;

  const _Float16* aptr  = &smemA[lrow * D_ + koff];                      // LDS
  const _Float16* bbase = ehat + (size_t)(tile_n * 64 + lrow) * D_ + koff;

  v8f acc0 = {}, acc1 = {}, acc2 = {}, acc3 = {};
  Frag16 a, b0, b1, b2, b3, an, b0n, b1n, b2n, b3n;

  // prologue: k0 = 0 fragments
  a.h[0]  = *(const v8h*)(aptr);              a.h[1]  = *(const v8h*)(aptr + 16);
  b0.h[0] = *(const v8h*)(bbase);             b0.h[1] = *(const v8h*)(bbase + 16);
  b1.h[0] = *(const v8h*)(bbase +  8192);     b1.h[1] = *(const v8h*)(bbase +  8192 + 16);
  b2.h[0] = *(const v8h*)(bbase + 16384);     b2.h[1] = *(const v8h*)(bbase + 16384 + 16);
  b3.h[0] = *(const v8h*)(bbase + 24576);     b3.h[1] = *(const v8h*)(bbase + 24576 + 16);

#pragma unroll
  for (int k0 = 0; k0 < D_; k0 += 32) {
    int kn = k0 + 32;
    if (kn < D_) {
      __builtin_prefetch(bbase + kn + 96, 0, 1);            // global_prefetch_b8
      __builtin_prefetch(bbase + 16384 + kn + 96, 0, 1);
      an.h[0]  = *(const v8h*)(aptr + kn);          an.h[1]  = *(const v8h*)(aptr + kn + 16);
      b0n.h[0] = *(const v8h*)(bbase + kn);         b0n.h[1] = *(const v8h*)(bbase + kn + 16);
      b1n.h[0] = *(const v8h*)(bbase +  8192 + kn); b1n.h[1] = *(const v8h*)(bbase +  8192 + kn + 16);
      b2n.h[0] = *(const v8h*)(bbase + 16384 + kn); b2n.h[1] = *(const v8h*)(bbase + 16384 + kn + 16);
      b3n.h[0] = *(const v8h*)(bbase + 24576 + kn); b3n.h[1] = *(const v8h*)(bbase + 24576 + kn + 16);
    }
    acc0 = __builtin_amdgcn_wmma_f32_16x16x32_f16(false, a.v, false, b0.v, (short)0, acc0, false, false);
    acc1 = __builtin_amdgcn_wmma_f32_16x16x32_f16(false, a.v, false, b1.v, (short)0, acc1, false, false);
    acc2 = __builtin_amdgcn_wmma_f32_16x16x32_f16(false, a.v, false, b2.v, (short)0, acc2, false, false);
    acc3 = __builtin_amdgcn_wmma_f32_16x16x32_f16(false, a.v, false, b3.v, (short)0, acc3, false, false);
    a = an; b0 = b0n; b1 = b1n; b2 = b2n; b3 = b3n;
  }

  // C/D layout: lane<16 -> N=lane, M=r (VGPR r); lane>=16 -> N=lane-16, M=r+8.
  int rbase = tile_m * 16 + ((lane < 16) ? 0 : 8);
  int cbase = tile_n * 64 + (lane & 15);
#pragma unroll
  for (int j = 0; j < 4; ++j) {
    int col = cbase + j * 16;
    if (col < V_) {
      float* out = sim + (size_t)rbase * V_ + col;
      v8f acc = (j == 0) ? acc0 : (j == 1) ? acc1 : (j == 2) ? acc2 : acc3;
#pragma unroll
      for (int r = 0; r < 8; ++r) out[(size_t)r * V_] = acc[r];
    }
  }
}

// ---------------------------------------------------------------------------
// Pass 4: tie-aware argmax per token. Order: sim desc, noise desc, index asc —
// matches max + (sim >= max) + argmax(where(tied, noise, -1)) exactly.
// ---------------------------------------------------------------------------
__global__ void __launch_bounds__(256)
tie_argmax(const float* __restrict__ sim, const float* __restrict__ noise,
           float* __restrict__ out_idx) {
  int t = blockIdx.x;
  const float* srow = sim   + (size_t)t * V_;
  const float* nrow = noise + (size_t)t * V_;

  float bs = -2.0f, bn = -2.0f; int bi = 0;
  for (int v = threadIdx.x; v < V_; v += blockDim.x) {
    float s = srow[v], n = nrow[v];
    bool better = (s > bs) || (s == bs && (n > bn || (n == bn && v < bi)));
    if (better) { bs = s; bn = n; bi = v; }
  }

  __shared__ float ss_[256];
  __shared__ float sn_[256];
  __shared__ int   si_[256];
  ss_[threadIdx.x] = bs; sn_[threadIdx.x] = bn; si_[threadIdx.x] = bi;
  __syncthreads();
  for (int off = 128; off > 0; off >>= 1) {
    if ((int)threadIdx.x < off) {
      float s = ss_[threadIdx.x + off], n = sn_[threadIdx.x + off];
      int   i = si_[threadIdx.x + off];
      float S = ss_[threadIdx.x],       N = sn_[threadIdx.x];
      int   I = si_[threadIdx.x];
      bool better = (s > S) || (s == S && (n > N || (n == N && i < I)));
      if (better) { ss_[threadIdx.x] = s; sn_[threadIdx.x] = n; si_[threadIdx.x] = i; }
    }
    __syncthreads();
  }
  if (threadIdx.x == 0) out_idx[t] = (float)si_[0];
}

// ---------------------------------------------------------------------------
extern "C" void kernel_launch(void* const* d_in, const int* in_sizes, int n_in,
                              void* d_out, int out_size, void* d_ws, size_t ws_size,
                              hipStream_t stream) {
  const int*   utt   = (const int*)  d_in[0];   // [B,S] int32
  const float* emb   = (const float*)d_in[1];   // [V,D] f32
  const float* grad  = (const float*)d_in[2];   // [B,S,D] f32
  const float* noise = (const float*)d_in[3];   // [B,S,V] f32

  float* out      = (float*)d_out;
  float* new_utt  = out;            // T_ values (indices as float)
  float* sim      = out + T_;       // T_ * V_ values

  _Float16* phat = (_Float16*)d_ws;                                   // 512 KB
  _Float16* ehat = (_Float16*)((char*)d_ws + (size_t)T_ * D_ * 2);    // ~50 MiB

  // Pass 1: normalize vocab table -> f16 (NPAD rows, one wave each)
  normalize_rows_f16<<<(NPAD * 32) / 256, 256, 0, stream>>>(emb, ehat, V_, NPAD);

  // Pass 2: gather + FGSM + normalize -> f16 (T_ rows)
  perturb_normalize_f16<<<(T_ * 32) / 256, 256, 0, stream>>>(utt, emb, grad, phat);

  // Pass 3: WMMA GEMM: 32 M-tiles x 99 groups of 8 N-tiles, 8 waves/block
  sim_gemm<<<32 * (NPAD / 64 / 8), 256, 0, stream>>>(phat, ehat, sim);

  // Pass 4: tie-aware argmax, one block per token
  tie_argmax<<<T_, 256, 0, stream>>>(sim, noise, new_utt);
}
